// SpanBertSboHead_35476429865143
// MI455X (gfx1250) — matrix operand order repaired and checked by reading, not compiled, not run on hardware
//
#include <hip/hip_runtime.h>
#include <hip/hip_bf16.h>

// SpanBERT SBO head for MI455X (gfx1250, wave32, WMMA).
// B=4, S=512, H=768, V=30522. Dominant op: [2048x768]x[768x30522] GEMM.
// bf16 WMMA (v_wmma_f32_16x16x32_bf16) with fp32 accumulate; weights are
// converted to bf16 padded to a 256-column multiple so the GEMM hot loop has
// zero bounds checks. A-tile staging uses the CDNA5 async global->LDS engine
// (global_load_async_to_lds_b128, ASYNCcnt); B-tile staging does a register
// k-pair pack so fragment reads are contiguous ds_load_b128. W leading
// dimension is a template constant so B row loads use immediate offsets.

typedef __bf16 bf16_t;
typedef bf16_t v16bf __attribute__((ext_vector_type(16)));
typedef float  v8f   __attribute__((ext_vector_type(8)));

#define BDIM   4
#define SDIM   512
#define HDIM   768
#define H3DIM  2304
#define VDIM   30522
#define VPAD   30720          // 120 * 256
#define MROWS  (BDIM * SDIM)  // 2048

// ---------------------------------------------------------------- scans ----
__global__ void span_scan_kernel(const int* __restrict__ span,
                                 int* __restrict__ fw, int* __restrict__ bw) {
  int b = blockIdx.x * blockDim.x + threadIdx.x;
  if (b >= BDIM) return;
  const int* sm = span + b * SDIM;
  int cur = 0;
  bool pm = sm[0] > -1;
  fw[b * SDIM + 0] = 0;
  for (int j = 1; j < SDIM; ++j) {
    bool m = sm[j] > -1;
    if (m && !pm) cur = j - 1;
    fw[b * SDIM + j] = cur;
    pm = m;
  }
  int nb = SDIM - 1;
  bool nm = sm[SDIM - 1] > -1;
  bw[b * SDIM + (SDIM - 1)] = SDIM - 1;
  for (int j = SDIM - 2; j >= 0; --j) {
    bool m = sm[j] > -1;
    if (m && !nm) nb = j + 1;
    bw[b * SDIM + j] = nb;
    nm = m;
  }
}

// --------------------------------------- f32 -> bf16 with column padding ----
__global__ __launch_bounds__(256) void cvt_pad_bf16_kernel(
    const float* __restrict__ in, bf16_t* __restrict__ out,
    int N, int Npad, int total) {
  int idx = blockIdx.x * 256 + threadIdx.x;
  if (idx >= total) return;
  int n = idx % Npad;
  int k = idx / Npad;
  out[idx] = (n < N) ? (bf16_t)in[(size_t)k * N + n] : (bf16_t)0.0f;
}

// ------------------------------------------------- sbo gather + concat ----
__global__ __launch_bounds__(256) void build_sbo_kernel(
    const float* __restrict__ hidden, const int* __restrict__ span,
    const float* __restrict__ pos_emb, const int* __restrict__ fw,
    const int* __restrict__ bw, bf16_t* __restrict__ sbo) {
  int idx = blockIdx.x * 256 + threadIdx.x;   // [0, 2048*2304)
  int c  = idx % H3DIM;
  int bs = idx / H3DIM;
  int s  = bs & (SDIM - 1);
  bool m = span[bs] > -1;
  float v = 0.0f;
  if (m) {
    int b = bs >> 9;
    if (c < HDIM) {
      v = hidden[((size_t)b * SDIM + fw[bs]) * HDIM + c];
    } else if (c < 2 * HDIM) {
      v = hidden[((size_t)b * SDIM + bw[bs]) * HDIM + (c - HDIM)];
    } else {
      v = pos_emb[(size_t)s * HDIM + (c - 2 * HDIM)];
    }
  }
  sbo[idx] = (bf16_t)v;
}

// ------------------------------------------------------- bf16 WMMA GEMM ----
// C[M,N] = A[M,K] @ W[K,LDW] (+ bias[N]).  LDW is a compile-time multiple of
// BN and padded W columns are zero, so the hot loop has no bounds checks and
// B row loads fold into immediate offsets. Block tile 64x256, 8 waves; each
// wave owns a 32x64 sub-tile (2x4 WMMA accumulators), K-step 32.
#define BM 64
#define BN 256
#define BK 32
#define ASTR 40     // A LDS stride in bf16 elems (80 B, 16B aligned)
#define BSTR 20     // B LDS stride in packed dwords (80 B, 16B aligned)

template <int LDW>
__global__ __launch_bounds__(256) void gemm_bf16_kernel(
    const bf16_t* __restrict__ A, const bf16_t* __restrict__ W,
    const float* __restrict__ bias, float* __restrict__ C,
    int M, int N, int K) {
  __shared__ bf16_t  As[BM * ASTR];      // [m][k] row-major
  __shared__ unsigned Bs[BN * BSTR];     // [n][kpair]: dword = (K=2j, K=2j+1)

  const int tid  = threadIdx.x;
  const int wave = tid >> 5;
  const int lane = tid & 31;
  const int m0 = blockIdx.y * BM;
  const int n0 = blockIdx.x * BN;
  const int mh = wave >> 2;   // 0..1 -> +32*mh rows
  const int nq = wave & 3;    // 0..3 -> +64*nq cols
  const int lr = lane & 15;
  const int kh = lane >> 4;   // 0/1

  // Staging maps
  const int arow = tid >> 2;            // 0..63
  const int acol = (tid & 3) * 8;       // 0,8,16,24
  const int kpb  = (tid >> 6) * 4;      // 0,4,8,12  (k-pair base)
  const int nb4  = (tid & 63) * 4;      // 0..252    (4 columns)

  v8f acc[2][4];
#pragma unroll
  for (int i = 0; i < 2; ++i)
#pragma unroll
    for (int j = 0; j < 4; ++j)
#pragma unroll
      for (int r = 0; r < 8; ++r) acc[i][j][r] = 0.0f;

  const bf16_t* ap = A + (size_t)(m0 + arow) * K + acol;
  const bf16_t* bp = W + (size_t)(2 * kpb) * LDW + n0 + nb4;

  // Wave-relative LDS byte offset of this thread's A staging slot
  // (flat AS3 address carries the LDS offset in bits [31:0]).
  const unsigned a_lds = (unsigned)(uintptr_t)(&As[arow * ASTR + acol]);

  for (int k0 = 0; k0 < K; k0 += BK) {
    // ---- stage A tile 64x32 via async DMA: 16B/lane global -> LDS ----
    asm volatile("global_load_async_to_lds_b128 %0, %1, off"
                 :: "v"(a_lds), "v"(ap) : "memory");

    // ---- stage B tile 32x256 transposed+packed: 8x b64 loads (immediate
    //      row offsets r*LDW), 4x b128 LDS stores of packed k-pair dwords.
    union Row { uint2 v; unsigned short u[4]; } rows[8];
#pragma unroll
    for (int r = 0; r < 8; ++r)
      rows[r].v = *reinterpret_cast<const uint2*>(bp + (size_t)r * LDW);
#pragma unroll
    for (int ni = 0; ni < 4; ++ni) {
      uint4 d;
      d.x = (unsigned)rows[0].u[ni] | ((unsigned)rows[1].u[ni] << 16);
      d.y = (unsigned)rows[2].u[ni] | ((unsigned)rows[3].u[ni] << 16);
      d.z = (unsigned)rows[4].u[ni] | ((unsigned)rows[5].u[ni] << 16);
      d.w = (unsigned)rows[6].u[ni] | ((unsigned)rows[7].u[ni] << 16);
      *reinterpret_cast<uint4*>(&Bs[(nb4 + ni) * BSTR + kpb]) = d;
    }

    // Prefetch next K-step's lines (global_prefetch_b8).
    if (k0 + BK < K) {
      __builtin_prefetch(ap + BK, 0, 1);
      __builtin_prefetch(bp + (size_t)BK * LDW, 0, 1);
    }

    // Async A copy must land in LDS before the barrier releases readers.
    asm volatile("s_wait_asynccnt 0x0" ::: "memory");
    __syncthreads();

    // ---- fragments per ISA 7.12.2 wave32 layouts ----
    v16bf afrag[2], bfrag[4];
#pragma unroll
    for (int mt = 0; mt < 2; ++mt) {
      int row = 32 * mh + 16 * mt + lr;     // M = lane%16
      uint4* p = reinterpret_cast<uint4*>(&afrag[mt]);
      p[0] = *reinterpret_cast<const uint4*>(&As[row * ASTR + kh * 8]);
      p[1] = *reinterpret_cast<const uint4*>(&As[row * ASTR + 16 + kh * 8]);
    }
#pragma unroll
    for (int nt = 0; nt < 4; ++nt) {
      int n = 64 * nq + 16 * nt + lr;       // N = lane%16
      uint4* p = reinterpret_cast<uint4*>(&bfrag[nt]);
      p[0] = *reinterpret_cast<const uint4*>(&Bs[n * BSTR + kh * 4]);
      p[1] = *reinterpret_cast<const uint4*>(&Bs[n * BSTR + 8 + kh * 4]);
    }
#pragma unroll
    for (int mt = 0; mt < 2; ++mt)
#pragma unroll
      for (int nt = 0; nt < 4; ++nt)
        acc[mt][nt] = __builtin_amdgcn_wmma_f32_16x16x32_bf16(
            false, afrag[mt], false, bfrag[nt], (short)0, acc[mt][nt],
            false, false);

    __syncthreads();

    ap += BK;
    bp += (size_t)BK * LDW;
  }

  // Epilogue: lane L -> N = L%16, VGPR r -> M = 8*(L/16)+r.
#pragma unroll
  for (int mt = 0; mt < 2; ++mt) {
#pragma unroll
    for (int nt = 0; nt < 4; ++nt) {
      int col = n0 + 64 * nq + 16 * nt + lr;
      if (col < N) {
        float badd = bias ? bias[col] : 0.0f;
#pragma unroll
        for (int r = 0; r < 8; ++r) {
          int row = m0 + 32 * mh + 16 * mt + 8 * kh + r;
          C[(size_t)row * N + col] = acc[mt][nt][r] + badd;
        }
      }
    }
  }
}

// ---------------------------------------- fused bias + gelu + LayerNorm ----
__global__ __launch_bounds__(256) void gelu_ln_kernel(
    const float* __restrict__ Cin, const float* __restrict__ bias,
    const float* __restrict__ gamma, const float* __restrict__ beta,
    bf16_t* __restrict__ out) {
  __shared__ float red[256];
  const int row = blockIdx.x;
  const int tid = threadIdx.x;
  const float* x = Cin + (size_t)row * HDIM;

  float y[3];
  float s = 0.0f;
#pragma unroll
  for (int i = 0; i < 3; ++i) {
    int h = tid + i * 256;
    float v = x[h] + bias[h];
    float g = 0.5f * v * (1.0f + erff(v * 0.70710678118654752f));
    y[i] = g;
    s += g;
  }
  red[tid] = s;
  __syncthreads();
  for (int off = 128; off > 0; off >>= 1) {
    if (tid < off) red[tid] += red[tid + off];
    __syncthreads();
  }
  float mu = red[0] * (1.0f / HDIM);
  __syncthreads();

  float vs = 0.0f;
#pragma unroll
  for (int i = 0; i < 3; ++i) {
    float d = y[i] - mu;
    vs += d * d;
  }
  red[tid] = vs;
  __syncthreads();
  for (int off = 128; off > 0; off >>= 1) {
    if (tid < off) red[tid] += red[tid + off];
    __syncthreads();
  }
  float inv = rsqrtf(red[0] * (1.0f / HDIM) + 1e-12f);

#pragma unroll
  for (int i = 0; i < 3; ++i) {
    int h = tid + i * 256;
    out[(size_t)row * HDIM + h] = (bf16_t)((y[i] - mu) * inv * gamma[h] + beta[h]);
  }
}

// ------------------------------------------------------------- launcher ----
extern "C" void kernel_launch(void* const* d_in, const int* in_sizes, int n_in,
                              void* d_out, int out_size, void* d_ws, size_t ws_size,
                              hipStream_t stream) {
  const float* hidden  = (const float*)d_in[0];
  const int*   span    = (const int*)d_in[1];
  const float* pos_emb = (const float*)d_in[2];
  const float* W1      = (const float*)d_in[3];
  const float* b1      = (const float*)d_in[4];
  const float* g1      = (const float*)d_in[5];
  const float* be1     = (const float*)d_in[6];
  const float* W2      = (const float*)d_in[7];
  const float* b2      = (const float*)d_in[8];
  const float* g2      = (const float*)d_in[9];
  const float* be2     = (const float*)d_in[10];
  const float* Wdec    = (const float*)d_in[11];
  const float* dbias   = (const float*)d_in[12];
  float* logits = (float*)d_out;

  // Workspace carve-up (all chunks 256B-aligned).
  char* base = (char*)d_ws;
  size_t off = 0;
  int* fw = (int*)(base + off);                  off += MROWS * sizeof(int);
  int* bw = (int*)(base + off);                  off += MROWS * sizeof(int);
  bf16_t* W1b  = (bf16_t*)(base + off);          off += (size_t)H3DIM * HDIM * 2;
  bf16_t* W2b  = (bf16_t*)(base + off);          off += (size_t)HDIM * HDIM * 2;
  bf16_t* Wdb  = (bf16_t*)(base + off);          off += (size_t)HDIM * VPAD * 2;
  bf16_t* sbo  = (bf16_t*)(base + off);          off += (size_t)MROWS * H3DIM * 2;
  float*  Cbuf = (float*)(base + off);           off += (size_t)MROWS * HDIM * 4;
  bf16_t* midb = (bf16_t*)(base + off);          off += (size_t)MROWS * HDIM * 2;
  bf16_t* preb = (bf16_t*)(base + off);          off += (size_t)MROWS * HDIM * 2;
  (void)ws_size; (void)n_in; (void)in_sizes; (void)out_size;

  // 1) span scans
  span_scan_kernel<<<1, 32, 0, stream>>>(span, fw, bw);

  // 2) weight conversions to bf16 (decoder padded to 30720 cols of zeros)
  {
    int n1 = H3DIM * HDIM;
    int n2 = HDIM * HDIM;
    int n3 = HDIM * VPAD;
    cvt_pad_bf16_kernel<<<(n1 + 255) / 256, 256, 0, stream>>>(W1, W1b, HDIM, HDIM, n1);
    cvt_pad_bf16_kernel<<<(n2 + 255) / 256, 256, 0, stream>>>(W2, W2b, HDIM, HDIM, n2);
    cvt_pad_bf16_kernel<<<(n3 + 255) / 256, 256, 0, stream>>>(Wdec, Wdb, VDIM, VPAD, n3);
  }

  // 3) gather + concat + mask -> sbo (bf16)
  build_sbo_kernel<<<(MROWS * H3DIM) / 256, 256, 0, stream>>>(
      hidden, span, pos_emb, fw, bw, sbo);

  // 4) GEMM1: [2048x2304] @ [2304x768] -> Cbuf
  gemm_bf16_kernel<HDIM><<<dim3(HDIM / BN, MROWS / BM), 256, 0, stream>>>(
      sbo, W1b, nullptr, Cbuf, MROWS, HDIM, H3DIM);
  // 5) gelu + LN -> mid (bf16)
  gelu_ln_kernel<<<MROWS, 256, 0, stream>>>(Cbuf, b1, g1, be1, midb);

  // 6) GEMM2: [2048x768] @ [768x768] -> Cbuf
  gemm_bf16_kernel<HDIM><<<dim3(HDIM / BN, MROWS / BM), 256, 0, stream>>>(
      midb, W2b, nullptr, Cbuf, MROWS, HDIM, HDIM);
  // 7) gelu + LN -> pre (bf16)
  gelu_ln_kernel<<<MROWS, 256, 0, stream>>>(Cbuf, b2, g2, be2, preb);

  // 8) Decoder GEMM: [2048x768] @ [768x30720pad] + bias -> logits (fp32)
  gemm_bf16_kernel<VPAD><<<dim3(VPAD / BN, MROWS / BM), 256, 0, stream>>>(
      preb, Wdb, dbias, logits, MROWS, VDIM, HDIM);
}
